// CustomWavLMAttention_85787676770575
// MI455X (gfx1250) — matrix operand order repaired and checked by reading, not compiled, not run on hardware
//
#include <hip/hip_runtime.h>
#include <hip/hip_bf16.h>

#define Bsz 4
#define Tq  1024
#define Ee  768
#define Hh  12
#define HDm 64
#define Mrows (Bsz * Tq)        // 4096
#define SROW 1032               // padded LDS row stride (floats)

typedef __bf16 bf16_t;
typedef __attribute__((ext_vector_type(16))) __bf16 v16bf;
typedef __attribute__((ext_vector_type(8)))  __bf16 v8bf;
typedef __attribute__((ext_vector_type(8)))  float  v8f;

// Build a 16-bit A-fragment (16x32) for wave32 WMMA.
// Lanes 0-15 hold row (lane), K = {k0..k0+15};
// lanes 16-31 hold row (lane-16), K = {k0+16..k0+31}.
__device__ __forceinline__ v16bf mk_afrag(const bf16_t* p) {
    v8bf lo = *(const v8bf*)(p);
    v8bf hi = *(const v8bf*)(p + 8);
    v16bf a;
#pragma unroll
    for (int i = 0; i < 8; ++i) { a[i] = lo[i]; a[i + 8] = hi[i]; }
    return a;
}

// ---------------------------------------------------------------------------
// Generic bf16 WMMA GEMM: OUT = ((A @ W^T) + bias [+ lora]) * scale
// A: (Mrows x Ee) bf16 row-major.  W: (Ee x Ee) bf16 row-major ([out][in]).
// Each wave computes a 32x64 output tile (two 16-row blocks sharing each
// B-fragment -> ~21 FLOP per fetched byte). MODE: 0 = bf16 row-major,
// 2 = bf16 transposed-V layout (B,H,HD,T), 3 = f32 row-major.
// ---------------------------------------------------------------------------
template <int MODE, bool LORA>
__global__ __launch_bounds__(256)
void gemm_wmma(const bf16_t* __restrict__ A, const bf16_t* __restrict__ W,
               const float* __restrict__ bias,
               const float* __restrict__ loraVec,   // (Mrows x 2) f32
               const float* __restrict__ loraB,     // (Ee x 2)    f32
               void* __restrict__ outp, float scale) {
    const int lane = threadIdx.x & 31;
    const int wave = threadIdx.x >> 5;
    const int wid  = blockIdx.x * 8 + wave;
    const int NT   = Ee / 64;                 // 12 N-strips
    const int mt   = wid / NT;
    const int nt   = wid % NT;
    if (mt >= Mrows / 32) return;
    const int m0 = mt * 32, n0 = nt * 64;

    v8f acc[2][4] = {};
    const bf16_t* Arow0 = A + (size_t)(m0 + (lane & 15)) * Ee;
    const bf16_t* Arow1 = Arow0 + (size_t)16 * Ee;
    const int aSel = (lane >= 16) ? 16 : 0;
    const int bSel = (lane >= 16) ? 16 : 0;

    for (int k0 = 0; k0 < Ee; k0 += 32) {
        v16bf a0 = mk_afrag(Arow0 + k0 + aSel);
        v16bf a1 = mk_afrag(Arow1 + k0 + aSel);
#pragma unroll
        for (int j = 0; j < 4; ++j) {
            const bf16_t* Bcol = W + (size_t)(n0 + j * 16 + (lane & 15)) * Ee;
            v16bf b = *(const v16bf*)(Bcol + k0 + bSel);
            acc[0][j] = __builtin_amdgcn_wmma_f32_16x16x32_bf16(
                false, a0, false, b, (short)0, acc[0][j], false, false);
            acc[1][j] = __builtin_amdgcn_wmma_f32_16x16x32_bf16(
                false, a1, false, b, (short)0, acc[1][j], false, false);
        }
    }

    const int rOff = (lane >= 16) ? 8 : 0;
#pragma unroll
    for (int half = 0; half < 2; ++half) {
#pragma unroll
        for (int r = 0; r < 8; ++r) {
            const int m = m0 + half * 16 + r + rOff;
            float l0 = 0.f, l1 = 0.f;
            if (LORA) { l0 = loraVec[m * 2 + 0]; l1 = loraVec[m * 2 + 1]; }
#pragma unroll
            for (int j = 0; j < 4; ++j) {
                const int col = n0 + j * 16 + (lane & 15);
                float v = acc[half][j][r] + bias[col];
                if (LORA) v += (l0 * loraB[col * 2 + 0] + l1 * loraB[col * 2 + 1]) * 0.5f;
                v *= scale;
                if (MODE == 0) {
                    ((bf16_t*)outp)[(size_t)m * Ee + col] = (bf16_t)v;
                } else if (MODE == 2) {           // V^T: (B,H,HD,T)
                    const int bb = m >> 10, t = m & 1023;
                    const int h  = col >> 6, d = col & 63;
                    ((bf16_t*)outp)[(((size_t)(bb * Hh + h)) * HDm + d) * Tq + t] = (bf16_t)v;
                } else {
                    ((float*)outp)[(size_t)m * Ee + col] = v;
                }
            }
        }
    }
}

// ---------------------------------------------------------------------------
// Attention: one block = (b, h, 64 query rows); 4 waves, 264 KB LDS scores.
// ---------------------------------------------------------------------------
__global__ __launch_bounds__(128)
void attn_kernel(const bf16_t* __restrict__ qb, const bf16_t* __restrict__ kb,
                 const bf16_t* __restrict__ vT, const float* __restrict__ gate,
                 const float* __restrict__ pbias, bf16_t* __restrict__ ctxb) {
    extern __shared__ float smem[];          // 64*SROW scores + 64 row sums
    float* S    = smem;
    float* lsum = smem + 64 * SROW;

    const int lane = threadIdx.x & 31;
    const int wave = threadIdx.x >> 5;
    const int bh   = blockIdx.x >> 4;
    const int qblk = blockIdx.x & 15;
    const int b    = bh / Hh, h = bh % Hh;
    const int q0   = qblk * 64;

    const int aSel = (lane >= 16) ? 16 : 0;
    const int bSel = (lane >= 16) ? 16 : 0;
    const int rOff = (lane >= 16) ? 8 : 0;

    // ----- Phase 1: S = Q K^T + gate * pos_bias  (per wave: 16 q-rows) -----
    const bf16_t* qrow =
        qb + ((size_t)(b * Tq) + q0 + wave * 16 + (lane & 15)) * Ee + h * HDm;
    v16bf a0 = mk_afrag(qrow + 0  + aSel);
    v16bf a1 = mk_afrag(qrow + 32 + aSel);

    float gv[8];
    const float* gbase = gate + ((size_t)(b * Hh + h)) * Tq + q0 + wave * 16;
#pragma unroll
    for (int r = 0; r < 8; ++r) gv[r] = gbase[r + rOff];
    const float* pbBase =
        pbias + (size_t)h * Tq * Tq + (size_t)(q0 + wave * 16) * Tq;

    for (int kt = 0; kt < Tq / 16; ++kt) {
        const bf16_t* krow =
            kb + ((size_t)(b * Tq) + kt * 16 + (lane & 15)) * Ee + h * HDm;
        v16bf b0 = *(const v16bf*)(krow + 0  + bSel);
        v16bf b1 = *(const v16bf*)(krow + 32 + bSel);
        v8f c = {};
        c = __builtin_amdgcn_wmma_f32_16x16x32_bf16(false, a0, false, b0, (short)0, c, false, false);
        c = __builtin_amdgcn_wmma_f32_16x16x32_bf16(false, a1, false, b1, (short)0, c, false, false);
        const int col = kt * 16 + (lane & 15);
#pragma unroll
        for (int r = 0; r < 8; ++r) {
            const int rl = wave * 16 + r + rOff;
            S[rl * SROW + col] = c[r] + gv[r] * pbBase[(size_t)(r + rOff) * Tq + col];
        }
    }
    __syncthreads();

    // ----- Phase 2: row softmax, 2 threads/row, float4 LDS accesses -----
    {
        const int row  = threadIdx.x >> 1;
        const int half = threadIdx.x & 1;
        float* Sr = S + (size_t)row * SROW + half * 512;
        float mx = -3.4e38f;
        for (int j = 0; j < 512; j += 4) {
            float4 f = *(const float4*)(Sr + j);
            mx = fmaxf(mx, fmaxf(fmaxf(f.x, f.y), fmaxf(f.z, f.w)));
        }
        mx = fmaxf(mx, __shfl_xor(mx, 1, 32));
        float sum = 0.f;
        for (int j = 0; j < 512; j += 4) {
            float4 f = *(const float4*)(Sr + j);
            f.x = __expf(f.x - mx); f.y = __expf(f.y - mx);
            f.z = __expf(f.z - mx); f.w = __expf(f.w - mx);
            *(float4*)(Sr + j) = f;
            sum += (f.x + f.y) + (f.z + f.w);
        }
        sum += __shfl_xor(sum, 1, 32);
        if (half == 0) lsum[row] = sum;
    }
    __syncthreads();

    // ----- Phase 3: ctx = P V  (P from LDS, f32 -> bf16 in A-layout) -----
    v8f acc[4] = {};
    const float* Sr = S + (size_t)(wave * 16 + (lane & 15)) * SROW;
    const bf16_t* vbase = vT + ((size_t)(b * Hh + h)) * HDm * Tq;
    for (int k0 = 0; k0 < Tq; k0 += 32) {
        const int kk = k0 + aSel;
        float4 f0 = *(const float4*)(Sr + kk);
        float4 f1 = *(const float4*)(Sr + kk + 4);
        float4 f2 = *(const float4*)(Sr + kk + 8);
        float4 f3 = *(const float4*)(Sr + kk + 12);
        v16bf a;
        a[0]  = (__bf16)f0.x; a[1]  = (__bf16)f0.y; a[2]  = (__bf16)f0.z; a[3]  = (__bf16)f0.w;
        a[4]  = (__bf16)f1.x; a[5]  = (__bf16)f1.y; a[6]  = (__bf16)f1.z; a[7]  = (__bf16)f1.w;
        a[8]  = (__bf16)f2.x; a[9]  = (__bf16)f2.y; a[10] = (__bf16)f2.z; a[11] = (__bf16)f2.w;
        a[12] = (__bf16)f3.x; a[13] = (__bf16)f3.y; a[14] = (__bf16)f3.z; a[15] = (__bf16)f3.w;
#pragma unroll
        for (int j = 0; j < 4; ++j) {
            const bf16_t* vcol = vbase + (size_t)(j * 16 + (lane & 15)) * Tq;
            v16bf bb = *(const v16bf*)(vcol + k0 + bSel);
            acc[j] = __builtin_amdgcn_wmma_f32_16x16x32_bf16(
                false, a, false, bb, (short)0, acc[j], false, false);
        }
    }
#pragma unroll
    for (int r = 0; r < 8; ++r) {
        const int rl  = wave * 16 + r + rOff;
        const float inv = 1.0f / lsum[rl];
        bf16_t* orow = ctxb + ((size_t)(b * Tq) + q0 + rl) * Ee + h * HDm;
#pragma unroll
        for (int j = 0; j < 4; ++j)
            orow[j * 16 + (lane & 15)] = (bf16_t)(acc[j][r] * inv);
    }
}

// ---------------------------------------------------------------------------
// Small scalar kernels
// ---------------------------------------------------------------------------
__global__ void cvt_kernel(const float* __restrict__ in, bf16_t* __restrict__ out, int n) {
    int i = blockIdx.x * 256 + threadIdx.x;
    if (i < n) out[i] = (bf16_t)in[i];
}

__global__ void lora_kernel(const float* __restrict__ x,
                            const float* __restrict__ Aq, const float* __restrict__ Ak,
                            const float* __restrict__ Av,
                            float* __restrict__ lq, float* __restrict__ lk,
                            float* __restrict__ lv) {
    int idx = blockIdx.x * blockDim.x + threadIdx.x;   // Mrows * 6
    if (idx >= Mrows * 6) return;
    int m = idx / 6, q = idx % 6, which = q >> 1, r = q & 1;
    const float* Arow = (which == 0 ? Aq : which == 1 ? Ak : Av) + r * Ee;
    const float* xr = x + (size_t)m * Ee;
    float s = 0.f;
    for (int e = 0; e < Ee; ++e) s += xr[e] * Arow[e];
    float* o = (which == 0 ? lq : which == 1 ? lk : lv);
    o[m * 2 + r] = s;
}

__global__ void gate_kernel(const float* __restrict__ x, const float* __restrict__ Wg,
                            const float* __restrict__ bg, const float* __restrict__ gru,
                            float* __restrict__ gate) {
    int idx = blockIdx.x * blockDim.x + threadIdx.x;   // Bsz*Hh*Tq
    if (idx >= Bsz * Hh * Tq) return;
    int b = idx / (Hh * Tq), h = (idx / Tq) % Hh, t = idx % Tq;
    const float* g = x + ((size_t)(b * Tq) + t) * Ee + h * HDm;
    float s0 = 0.f, s1 = 0.f;
    for (int j = 0; j < 8; ++j) {
        float s = bg[j];
        const float* w = Wg + j * HDm;
        for (int d = 0; d < HDm; ++d) s += g[d] * w[d];
        if (j < 4) s0 += s; else s1 += s;
    }
    float ga = 1.f / (1.f + __expf(-s0));
    float gb = 1.f / (1.f + __expf(-s1));
    gate[idx] = ga * (gb * gru[h] - 1.f) + 2.f;        // idx == (b*H+h)*T + t
}

__global__ void posbias_kernel(const float* __restrict__ rel_embed, float* __restrict__ pb) {
    int idx = blockIdx.x * 256 + threadIdx.x;          // Tq*Tq
    if (idx >= Tq * Tq) return;
    int i = idx >> 10, j = idx & 1023;
    int rel = j - i;
    int bucket = (rel > 0) ? 160 : 0;
    int ar = (rel < 0) ? -rel : rel;
    int val;
    if (ar < 80) val = ar;
    else {
        float lr = logf((float)ar * (1.0f / 80.0f)) * (80.0f / 2.302585093f); // *(nb-80)/log(10)
        int large = 80 + (int)lr;
        val = (large < 159) ? large : 159;
    }
    bucket += val;
    const float* re = rel_embed + bucket * Hh;
#pragma unroll
    for (int h = 0; h < Hh; ++h) pb[(size_t)h * Tq * Tq + idx] = re[h];
}

// ---------------------------------------------------------------------------
extern "C" void kernel_launch(void* const* d_in, const int* in_sizes, int n_in,
                              void* d_out, int out_size, void* d_ws, size_t ws_size,
                              hipStream_t stream) {
    const float* x   = (const float*)d_in[0];
    const float* Wq  = (const float*)d_in[1];   const float* bq = (const float*)d_in[2];
    const float* Wk  = (const float*)d_in[3];   const float* bk = (const float*)d_in[4];
    const float* Wv  = (const float*)d_in[5];   const float* bv = (const float*)d_in[6];
    const float* Aq  = (const float*)d_in[7];   const float* Bq = (const float*)d_in[8];
    const float* Ak  = (const float*)d_in[9];   const float* Bk = (const float*)d_in[10];
    const float* Av  = (const float*)d_in[11];  const float* Bv = (const float*)d_in[12];
    const float* Wo  = (const float*)d_in[13];  const float* bo = (const float*)d_in[14];
    const float* Wg  = (const float*)d_in[15];  const float* bg = (const float*)d_in[16];
    const float* gru = (const float*)d_in[17];  const float* rel = (const float*)d_in[18];
    float* out = (float*)d_out;

    char* ws = (char*)d_ws;
    size_t off = 0;
    auto alloc = [&](size_t bytes) -> void* {
        void* p = ws + off;
        off += (bytes + 255) & ~(size_t)255;
        return p;
    };
    bf16_t* xb   = (bf16_t*)alloc((size_t)Mrows * Ee * 2);
    bf16_t* wqb  = (bf16_t*)alloc((size_t)Ee * Ee * 2);
    bf16_t* wkb  = (bf16_t*)alloc((size_t)Ee * Ee * 2);
    bf16_t* wvb  = (bf16_t*)alloc((size_t)Ee * Ee * 2);
    bf16_t* wob  = (bf16_t*)alloc((size_t)Ee * Ee * 2);
    bf16_t* q1b  = (bf16_t*)alloc((size_t)Mrows * Ee * 2);
    bf16_t* k1b  = (bf16_t*)alloc((size_t)Mrows * Ee * 2);
    bf16_t* v1b  = (bf16_t*)alloc((size_t)Mrows * Ee * 2);
    bf16_t* qbuf = (bf16_t*)alloc((size_t)Mrows * Ee * 2);
    bf16_t* kbuf = (bf16_t*)alloc((size_t)Mrows * Ee * 2);
    bf16_t* vTb  = (bf16_t*)alloc((size_t)Mrows * Ee * 2);
    bf16_t* ctxb = (bf16_t*)alloc((size_t)Mrows * Ee * 2);
    float*  lq   = (float*)alloc((size_t)Mrows * 2 * 4);
    float*  lk   = (float*)alloc((size_t)Mrows * 2 * 4);
    float*  lv   = (float*)alloc((size_t)Mrows * 2 * 4);
    float*  gate = (float*)alloc((size_t)Bsz * Hh * Tq * 4);
    float*  pb   = (float*)alloc((size_t)Hh * Tq * Tq * 4);

    int n;
    n = Mrows * Ee; cvt_kernel<<<(n + 255) / 256, 256, 0, stream>>>(x,  xb,  n);
    n = Ee * Ee;    cvt_kernel<<<(n + 255) / 256, 256, 0, stream>>>(Wq, wqb, n);
    cvt_kernel<<<(n + 255) / 256, 256, 0, stream>>>(Wk, wkb, n);
    cvt_kernel<<<(n + 255) / 256, 256, 0, stream>>>(Wv, wvb, n);
    cvt_kernel<<<(n + 255) / 256, 256, 0, stream>>>(Wo, wob, n);

    lora_kernel<<<(Mrows * 6 + 127) / 128, 128, 0, stream>>>(x, Aq, Ak, Av, lq, lk, lv);
    gate_kernel<<<(Bsz * Hh * Tq + 127) / 128, 128, 0, stream>>>(x, Wg, bg, gru, gate);
    posbias_kernel<<<(Tq * Tq + 255) / 256, 256, 0, stream>>>(rel, pb);

    const int gblocks = (Mrows / 32) * (Ee / 64) / 8;   // 192
    gemm_wmma<0, true ><<<gblocks, 256, 0, stream>>>(xb,  wqb, bq, lq, Bq, q1b, 1.0f);
    gemm_wmma<0, true ><<<gblocks, 256, 0, stream>>>(xb,  wkb, bk, lk, Bk, k1b, 1.0f);
    gemm_wmma<0, true ><<<gblocks, 256, 0, stream>>>(xb,  wvb, bv, lv, Bv, v1b, 1.0f);
    gemm_wmma<0, false><<<gblocks, 256, 0, stream>>>(q1b, wqb, bq, nullptr, nullptr, qbuf, 0.125f);
    gemm_wmma<0, false><<<gblocks, 256, 0, stream>>>(k1b, wkb, bk, nullptr, nullptr, kbuf, 1.0f);
    gemm_wmma<2, false><<<gblocks, 256, 0, stream>>>(v1b, wvb, bv, nullptr, nullptr, vTb, 1.0f);

    const size_t attnLds = (size_t)(64 * SROW + 64) * sizeof(float);  // ~264 KB (<320 KB WGP)
    attn_kernel<<<Bsz * Hh * (Tq / 64), 128, attnLds, stream>>>(qbuf, kbuf, vTb, gate, pb, ctxb);

    gemm_wmma<3, false><<<gblocks, 256, 0, stream>>>(ctxb, wob, bo, nullptr, nullptr, (void*)out, 1.0f);
}
